// EncoderAttention_74801150427279
// MI455X (gfx1250) — matrix operand order repaired and checked by reading, not compiled, not run on hardware
//
#include <hip/hip_runtime.h>

// ---------------------------------------------------------------------------
// EncoderAttention for MI455X (gfx1250): bf16 WMMA pipeline.
//   - ping-pong double-buffered fragments (loads land directly in loop-carried
//     registers; zero copy insertion)
//   - attention: row-max via shfl_xor, row-sum via an extra WMMA against a
//     constant ones-matrix (l = P x 1), softmax scale pre-folded into Q
//   1) cast hidden -> bf16 ; transpose-cast W_qkv, W_out -> [N][K] bf16
//   2) QKV GEMM -> Q*scale[B,H,T,D], K[B,H,T,D], V^T[B,H,D,T]
//   3) flash attention per (b,h)
//   4) out GEMM + bias -> f32 d_out
// Workspace required: ~92 MB.
// ---------------------------------------------------------------------------

typedef __bf16 bf16;
typedef __attribute__((ext_vector_type(8)))  bf16  v8bf;
typedef __attribute__((ext_vector_type(16))) bf16  v16bf;
typedef __attribute__((ext_vector_type(8)))  float v8f;

static constexpr int Bb   = 8;
static constexpr int T    = 1024;
static constexpr int C    = 1024;
static constexpr int H    = 16;
static constexpr int D    = 64;
static constexpr int NQKV = 3 * C;
static constexpr int M    = Bb * T;   // 8192

__device__ __forceinline__ v8f wmma_bf16(v16bf a, v16bf b, v8f c) {
  return __builtin_amdgcn_wmma_f32_16x16x32_bf16(false, a, false, b, (short)0, c,
                                                 false, false);
}

// A fragment (16x32, 16-bit): lane<16 -> row=lane, K={0..15};
// lane>=16 -> row=lane-16, K={16..31}. rowbase = row start + k0.
__device__ __forceinline__ v16bf load_a_frag(const bf16* rowbase, int lane) {
  int kb = (lane >> 4) * 16;
  v8bf lo = *(const v8bf*)(rowbase + kb);
  v8bf hi = *(const v8bf*)(rowbase + kb + 8);
  return __builtin_shufflevector(lo, hi, 0, 1, 2, 3, 4, 5, 6, 7,
                                 8, 9, 10, 11, 12, 13, 14, 15);
}

// B fragment (32x16, 16-bit): lane holds column lane&15; 16 consecutive K from
// (lane>>4)*16. p = &src[col][k0 + (lane>>4)*16] on a [N][K] source.
__device__ __forceinline__ v16bf load_b_frag(const bf16* p) {
  v8bf lo = *(const v8bf*)p;
  v8bf hi = *(const v8bf*)(p + 8);
  return __builtin_shufflevector(lo, hi, 0, 1, 2, 3, 4, 5, 6, 7,
                                 8, 9, 10, 11, 12, 13, 14, 15);
}

// ---------------------------------------------------------------------------
__global__ void k_cast_bf16(const float* __restrict__ src, bf16* __restrict__ dst,
                            int n) {
  int i = blockIdx.x * blockDim.x + threadIdx.x;
  int stride = gridDim.x * blockDim.x;
  for (; i < n; i += stride) dst[i] = (bf16)src[i];
}

// Wt[n][k] = W[k][n] as bf16. W is [K][N] row-major.
__global__ void k_transpose_bf16(const float* __restrict__ W, bf16* __restrict__ Wt,
                                 int K, int N) {
  int i = blockIdx.x * blockDim.x + threadIdx.x;
  int stride = gridDim.x * blockDim.x;
  int total = K * N;
  for (; i < total; i += stride) {
    int k = i / N, n = i - k * N;
    Wt[(size_t)n * K + k] = (bf16)W[i];
  }
}

// ---------------------------------------------------------------------------
// QKV GEMM: [8192][1024] x Wt[3072][1024] -> Q*scale / K / V^T (bf16).
// Wave owns a 32x64 tile; two fragment sets ping-pong.
__global__ __launch_bounds__(256) void k_qkv_gemm(
    const bf16* __restrict__ A, const bf16* __restrict__ Wt,
    const float* __restrict__ bias, bf16* __restrict__ Qo,
    bf16* __restrict__ Ko, bf16* __restrict__ Vt) {
  const int lane = threadIdx.x & 31;
  const int w    = threadIdx.x >> 5;
  const int mlo  = lane & 15;
  const int g    = lane >> 4;
  const int row0 = blockIdx.y * 256 + w * 32;
  const int col0 = blockIdx.x * 64;

  v8f acc[2][4];
#pragma unroll
  for (int rb = 0; rb < 2; rb++)
#pragma unroll
    for (int i = 0; i < 4; i++) acc[rb][i] = (v8f){0, 0, 0, 0, 0, 0, 0, 0};

  const bf16* arow0 = A + (size_t)(row0 + mlo) * C;
  const bf16* arow1 = A + (size_t)(row0 + 16 + mlo) * C;
  const bf16* wcol[4];
#pragma unroll
  for (int nt = 0; nt < 4; nt++)
    wcol[nt] = Wt + (size_t)(col0 + nt * 16 + mlo) * C + g * 16;

  // Set A covers kk, set B covers kk+32.
  v16bf a0A = load_a_frag(arow0, lane);
  v16bf a1A = load_a_frag(arow1, lane);
  v16bf bA[4];
#pragma unroll
  for (int nt = 0; nt < 4; nt++) bA[nt] = load_b_frag(wcol[nt]);
  v16bf a0B = load_a_frag(arow0 + 32, lane);
  v16bf a1B = load_a_frag(arow1 + 32, lane);
  v16bf bB[4];
#pragma unroll
  for (int nt = 0; nt < 4; nt++) bB[nt] = load_b_frag(wcol[nt] + 32);

  for (int kk = 0; kk < C; kk += 64) {
#pragma unroll
    for (int nt = 0; nt < 4; nt++) acc[0][nt] = wmma_bf16(a0A, bA[nt], acc[0][nt]);
#pragma unroll
    for (int nt = 0; nt < 4; nt++) acc[1][nt] = wmma_bf16(a1A, bA[nt], acc[1][nt]);
    int knA = (kk + 64) & (C - 1);   // wraps dead on last iter
    a0A = load_a_frag(arow0 + knA, lane);
    a1A = load_a_frag(arow1 + knA, lane);
#pragma unroll
    for (int nt = 0; nt < 4; nt++) bA[nt] = load_b_frag(wcol[nt] + knA);

#pragma unroll
    for (int nt = 0; nt < 4; nt++) acc[0][nt] = wmma_bf16(a0B, bB[nt], acc[0][nt]);
#pragma unroll
    for (int nt = 0; nt < 4; nt++) acc[1][nt] = wmma_bf16(a1B, bB[nt], acc[1][nt]);
    int knB = (kk + 96) & (C - 1);
    a0B = load_a_frag(arow0 + knB, lane);
    a1B = load_a_frag(arow1 + knB, lane);
#pragma unroll
    for (int nt = 0; nt < 4; nt++) bB[nt] = load_b_frag(wcol[nt] + knB);
  }

  const float qscale = 0.125f;   // 1/sqrt(D): folded into Q here
#pragma unroll
  for (int nt = 0; nt < 4; nt++) {
    int col = col0 + nt * 16 + mlo;
    float bs = bias[col];
    int sec = col >> 10;   // 0=Q, 1=K, 2=V
    int c2  = col & 1023;
    int h   = c2 >> 6;
    int d   = c2 & 63;
#pragma unroll
    for (int rb = 0; rb < 2; rb++) {
#pragma unroll
      for (int r = 0; r < 8; r++) {
        int row  = row0 + rb * 16 + g * 8 + r;
        int bidx = row >> 10;
        int t    = row & 1023;
        float v  = acc[rb][nt][r] + bs;
        size_t bh = (size_t)(bidx * H + h);
        if (sec == 0)      Qo[(bh * T + t) * D + d] = (bf16)(v * qscale);
        else if (sec == 1) Ko[(bh * T + t) * D + d] = (bf16)v;
        else               Vt[(bh * D + d) * T + t] = (bf16)v;   // transposed
      }
    }
  }
}

// ---------------------------------------------------------------------------
// Flash attention: grid.x = B*H, grid.y = T/128. Wave owns 16 query rows.
// Row-sum of exp computed as an extra WMMA (P x ones) -> no shuffle tree;
// only the row-max needs the 4-step shfl_xor reduction.
__global__ __launch_bounds__(256) void k_attention(
    const bf16* __restrict__ Q, const bf16* __restrict__ Kc,
    const bf16* __restrict__ Vt, bf16* __restrict__ O) {
  __shared__ __align__(16) bf16 pbuf[8][16 * 32];   // wave-private P tiles
  const int lane = threadIdx.x & 31;
  const int w    = threadIdx.x >> 5;
  const int mlo  = lane & 15;
  const int g    = lane >> 4;
  const int bh   = blockIdx.x;
  const int bidx = bh >> 4;
  const int h    = bh & 15;
  const int q0   = blockIdx.y * 128 + w * 16;

  const bf16* qbh = Q  + (size_t)bh * T * D;
  const bf16* kbh = Kc + (size_t)bh * T * D;
  const bf16* vbh = Vt + (size_t)bh * D * T;

  v16bf aq0 = load_a_frag(qbh + (size_t)(q0 + mlo) * D, lane);
  v16bf aq1 = load_a_frag(qbh + (size_t)(q0 + mlo) * D + 32, lane);

  v16bf onesb;
#pragma unroll
  for (int j = 0; j < 16; j++) onesb[j] = (bf16)1.0f;

  v8f o[4];
#pragma unroll
  for (int i = 0; i < 4; i++) o[i] = (v8f){0, 0, 0, 0, 0, 0, 0, 0};
  v8f lacc = (v8f){0, 0, 0, 0, 0, 0, 0, 0};   // per-row sum of exp (P x ones)
  float mi[8];
#pragma unroll
  for (int r = 0; r < 8; r++) mi[r] = -3.0e38f;

  // One 32-key chunk: V frags -> S wmmas -> online softmax -> P via LDS -> P*V.
  auto chunk = [&](int k0c, const v16bf* bk) {
    v16bf bv[4];
#pragma unroll
    for (int nt = 0; nt < 4; nt++)
      bv[nt] = load_b_frag(vbh + (size_t)(nt * 16 + mlo) * T + k0c + g * 16);

    v8f s0 = (v8f){0, 0, 0, 0, 0, 0, 0, 0};
    v8f s1 = (v8f){0, 0, 0, 0, 0, 0, 0, 0};
    s0 = wmma_bf16(aq0, bk[0], s0);
    s0 = wmma_bf16(aq1, bk[1], s0);
    s1 = wmma_bf16(aq0, bk[2], s1);
    s1 = wmma_bf16(aq1, bk[3], s1);

#pragma unroll
    for (int r = 0; r < 8; r++) {
      float x0 = s0[r];            // scale already folded into Q
      float x1 = s1[r];
      float cm = fmaxf(x0, x1);
#pragma unroll
      for (int msk = 8; msk >= 1; msk >>= 1)
        cm = fmaxf(cm, __shfl_xor(cm, msk, 16));
      float mnew  = fmaxf(mi[r], cm);
      float alpha = __expf(mi[r] - mnew);
      mi[r] = mnew;
      float p0 = __expf(x0 - mnew);
      float p1 = __expf(x1 - mnew);
      o[0][r] *= alpha; o[1][r] *= alpha; o[2][r] *= alpha; o[3][r] *= alpha;
      lacc[r] *= alpha;
      int rowm = r + g * 8;
      pbuf[w][rowm * 32 + mlo]      = (bf16)p0;
      pbuf[w][rowm * 32 + 16 + mlo] = (bf16)p1;
    }
    // Wave-private LDS; DS pipeline is in-order per wave.
    asm volatile("s_wait_dscnt 0x0" ::: "memory");
    v16bf ap = load_a_frag(&pbuf[w][mlo * 32], lane);
    lacc = wmma_bf16(ap, onesb, lacc);   // row-sum of exp, no cross-lane tree
#pragma unroll
    for (int nt = 0; nt < 4; nt++) o[nt] = wmma_bf16(ap, bv[nt], o[nt]);
  };

  auto load_k_set = [&](v16bf* bk, int k0c) {
    bk[0] = load_b_frag(kbh + (size_t)(k0c + mlo) * D + g * 16);
    bk[1] = load_b_frag(kbh + (size_t)(k0c + mlo) * D + 32 + g * 16);
    bk[2] = load_b_frag(kbh + (size_t)(k0c + 16 + mlo) * D + g * 16);
    bk[3] = load_b_frag(kbh + (size_t)(k0c + 16 + mlo) * D + 32 + g * 16);
  };

  v16bf bkA[4], bkB[4];
  load_k_set(bkA, 0);
  load_k_set(bkB, 32);

  for (int k0 = 0; k0 < T; k0 += 64) {
    chunk(k0, bkA);
    load_k_set(bkA, (k0 + 64) & (T - 1));   // dead on last iter
    chunk(k0 + 32, bkB);
    load_k_set(bkB, (k0 + 96) & (T - 1));
  }

#pragma unroll
  for (int r = 0; r < 8; r++) {
    float inv = 1.0f / lacc[r];
    int t = q0 + r + g * 8;
    size_t base = ((size_t)bidx * T + t) * C + h * D;
#pragma unroll
    for (int nt = 0; nt < 4; nt++)
      O[base + nt * 16 + mlo] = (bf16)(o[nt][r] * inv);
  }
}

// ---------------------------------------------------------------------------
// Output projection: [8192][1024] x Wt_out[1024][1024] + bias -> f32 d_out.
__global__ __launch_bounds__(256) void k_out_gemm(
    const bf16* __restrict__ A, const bf16* __restrict__ Wt,
    const float* __restrict__ bias, float* __restrict__ Y) {
  const int lane = threadIdx.x & 31;
  const int w    = threadIdx.x >> 5;
  const int mlo  = lane & 15;
  const int g    = lane >> 4;
  const int row0 = blockIdx.y * 256 + w * 32;
  const int col0 = blockIdx.x * 64;

  v8f acc[2][4];
#pragma unroll
  for (int rb = 0; rb < 2; rb++)
#pragma unroll
    for (int i = 0; i < 4; i++) acc[rb][i] = (v8f){0, 0, 0, 0, 0, 0, 0, 0};

  const bf16* arow0 = A + (size_t)(row0 + mlo) * C;
  const bf16* arow1 = A + (size_t)(row0 + 16 + mlo) * C;
  const bf16* wcol[4];
#pragma unroll
  for (int nt = 0; nt < 4; nt++)
    wcol[nt] = Wt + (size_t)(col0 + nt * 16 + mlo) * C + g * 16;

  v16bf a0A = load_a_frag(arow0, lane);
  v16bf a1A = load_a_frag(arow1, lane);
  v16bf bA[4];
#pragma unroll
  for (int nt = 0; nt < 4; nt++) bA[nt] = load_b_frag(wcol[nt]);
  v16bf a0B = load_a_frag(arow0 + 32, lane);
  v16bf a1B = load_a_frag(arow1 + 32, lane);
  v16bf bB[4];
#pragma unroll
  for (int nt = 0; nt < 4; nt++) bB[nt] = load_b_frag(wcol[nt] + 32);

  for (int kk = 0; kk < C; kk += 64) {
#pragma unroll
    for (int nt = 0; nt < 4; nt++) acc[0][nt] = wmma_bf16(a0A, bA[nt], acc[0][nt]);
#pragma unroll
    for (int nt = 0; nt < 4; nt++) acc[1][nt] = wmma_bf16(a1A, bA[nt], acc[1][nt]);
    int knA = (kk + 64) & (C - 1);
    a0A = load_a_frag(arow0 + knA, lane);
    a1A = load_a_frag(arow1 + knA, lane);
#pragma unroll
    for (int nt = 0; nt < 4; nt++) bA[nt] = load_b_frag(wcol[nt] + knA);

#pragma unroll
    for (int nt = 0; nt < 4; nt++) acc[0][nt] = wmma_bf16(a0B, bB[nt], acc[0][nt]);
#pragma unroll
    for (int nt = 0; nt < 4; nt++) acc[1][nt] = wmma_bf16(a1B, bB[nt], acc[1][nt]);
    int knB = (kk + 96) & (C - 1);
    a0B = load_a_frag(arow0 + knB, lane);
    a1B = load_a_frag(arow1 + knB, lane);
#pragma unroll
    for (int nt = 0; nt < 4; nt++) bB[nt] = load_b_frag(wcol[nt] + knB);
  }

#pragma unroll
  for (int nt = 0; nt < 4; nt++) {
    int col = col0 + nt * 16 + mlo;
    float bs = bias[col];
#pragma unroll
    for (int rb = 0; rb < 2; rb++)
#pragma unroll
      for (int r = 0; r < 8; r++) {
        int row = row0 + rb * 16 + g * 8 + r;
        Y[(size_t)row * C + col] = acc[rb][nt][r] + bs;
      }
  }
}

// ---------------------------------------------------------------------------
extern "C" void kernel_launch(void* const* d_in, const int* in_sizes, int n_in,
                              void* d_out, int out_size, void* d_ws, size_t ws_size,
                              hipStream_t stream) {
  (void)in_sizes; (void)n_in; (void)out_size; (void)ws_size;
  const float* hidden = (const float*)d_in[0];
  const float* W_qkv  = (const float*)d_in[1];
  const float* b_qkv  = (const float*)d_in[2];
  const float* W_out  = (const float*)d_in[3];
  const float* b_out  = (const float*)d_in[4];
  float* out = (float*)d_out;

  char* ws = (char*)d_ws;
  bf16* hb    = (bf16*)ws; ws += (size_t)M * C * 2;        // 16 MB
  bf16* wqkvt = (bf16*)ws; ws += (size_t)NQKV * C * 2;     //  6 MB
  bf16* woutt = (bf16*)ws; ws += (size_t)C * C * 2;        //  2 MB
  bf16* Qb    = (bf16*)ws; ws += (size_t)M * C * 2;        // 16 MB
  bf16* Kb    = (bf16*)ws; ws += (size_t)M * C * 2;        // 16 MB
  bf16* Vtb   = (bf16*)ws; ws += (size_t)M * C * 2;        // 16 MB
  bf16* attnb = (bf16*)ws; ws += (size_t)M * C * 2;        // 16 MB

  k_cast_bf16<<<2048, 256, 0, stream>>>(hidden, hb, M * C);
  k_transpose_bf16<<<2048, 256, 0, stream>>>(W_qkv, wqkvt, C, NQKV);
  k_transpose_bf16<<<1024, 256, 0, stream>>>(W_out, woutt, C, C);

  k_qkv_gemm<<<dim3(NQKV / 64, M / 256), 256, 0, stream>>>(hb, wqkvt, b_qkv,
                                                           Qb, Kb, Vtb);
  k_attention<<<dim3(Bb * H, T / 128), 256, 0, stream>>>(Qb, Kb, Vtb, attnb);
  k_out_gemm<<<dim3(C / 64, M / 256), 256, 0, stream>>>(attnb, woutt, b_out, out);
}
